// Attention_60730837566038
// MI455X (gfx1250) — matrix operand order repaired and checked by reading, not compiled, not run on hardware
//
#include <hip/hip_runtime.h>
#include <hip/hip_bf16.h>

typedef unsigned short u16;
typedef __attribute__((ext_vector_type(16))) __bf16 bf16x16;
typedef __attribute__((ext_vector_type(8)))  float  f32x8;

union Frag {
    uint4   u[2];
    u16     us[16];
    bf16x16 v;
};

__device__ __forceinline__ u16 f2bf(float x) {
    __bf16 b = (__bf16)x;
    return __builtin_bit_cast(unsigned short, b);
}

__device__ __forceinline__ f32x8 zero8() {
    f32x8 z;
#pragma unroll
    for (int i = 0; i < 8; ++i) z[i] = 0.0f;
    return z;
}

// A-matrix fragment (16x32 bf16, MxK), source row-major [row][k], leading dim ld.
// ISA layout: lanes 0-15 -> M=lane, K = k0+{0..7} and k0+{16..23}
//             lanes 16-31 -> M=lane-16, K = k0+{8..15} and k0+{24..31}
__device__ __forceinline__ Frag load_fragA(const u16* base, int row0, int ld, int k0, int lane) {
    Frag f;
    int r    = row0 + (lane & 15);
    int koff = (lane & 16) ? 8 : 0;
    const u16* p = base + (size_t)r * ld + k0 + koff;
    f.u[0] = *(const uint4*)(p);
    f.u[1] = *(const uint4*)(p + 16);
    return f;
}

// B-matrix fragment (32x16 bf16, KxN), source holds B^T row-major: src[col][k], leading dim ld.
// ISA layout: lanes 0-15 -> N=lane, K = k0+{0..7} and k0+{16..23} ; lanes 16-31 -> N=lane-16, K = k0+{8..15} and k0+{24..31}
__device__ __forceinline__ Frag load_fragB(const u16* base, int col0, int ld, int k0, int lane) {
    Frag f;
    int c    = col0 + (lane & 15);
    int koff = (lane & 16) ? 8 : 0;
    const u16* p = base + (size_t)c * ld + k0 + koff;
    f.u[0] = *(const uint4*)(p);
    f.u[1] = *(const uint4*)(p + 16);
    return f;
}

__device__ __forceinline__ f32x8 wmma_bf16(const Frag& a, const Frag& b, f32x8 c) {
    return __builtin_amdgcn_wmma_f32_16x16x32_bf16(
        /*neg_a=*/false, a.v, /*neg_b=*/false, b.v,
        /*c_mod=*/(short)0, c, /*reuse_a=*/false, /*reuse_b=*/false);
}

// ---------------------------------------------------------------------------
// Kernel 1: fp32 -> bf16 bulk convert (X), vectorized float4 -> ushort4
// ---------------------------------------------------------------------------
__global__ void cvt_f32_bf16(const float* __restrict__ in, u16* __restrict__ out, int n4) {
    int i      = blockIdx.x * blockDim.x + threadIdx.x;
    int stride = gridDim.x * blockDim.x;
    for (; i < n4; i += stride) {
        float4 f = ((const float4*)in)[i];
        ushort4 h;
        h.x = f2bf(f.x); h.y = f2bf(f.y); h.z = f2bf(f.z); h.w = f2bf(f.w);
        ((ushort4*)out)[i] = h;
    }
}

// ---------------------------------------------------------------------------
// Kernel 2: W (2048 x 256 fp32, row-major) -> Wt (256 x 2048 bf16)
// ---------------------------------------------------------------------------
__global__ void transpose_w(const float* __restrict__ W, u16* __restrict__ Wt) {
    int idx = blockIdx.x * blockDim.x + threadIdx.x;   // 524288 total
    int k = idx >> 8;
    int n = idx & 255;
    Wt[(size_t)n * 2048 + k] = f2bf(W[idx]);
}

// ---------------------------------------------------------------------------
// Kernel 3: C[8192 x 256] = Xb[8192 x 2048] * Wt^T  (Wt stored as [256 x 2048])
// One wave per 16x64 output tile, manually double-buffered: fragments for
// K-step k+32 are fetched before the 4 WMMAs of step k are issued.
// transposed=1 stores V as Vt[b][n][l].
// ---------------------------------------------------------------------------
__global__ void gemm_bf16(const u16* __restrict__ Xb, const u16* __restrict__ Wt,
                          u16* __restrict__ C, int transposed) {
    const int lane = threadIdx.x & 31;
    const int widx = blockIdx.x * (blockDim.x >> 5) + (threadIdx.x >> 5); // 0..2047
    const int mtile = widx >> 2;        // 512 m-tiles
    const int nq    = widx & 3;         // 4 n-quads of 64
    const int n0    = nq * 64;
    const int m0    = mtile * 16;

    f32x8 acc[4];
#pragma unroll
    for (int t = 0; t < 4; ++t) acc[t] = zero8();

    // Prologue: fetch k0 = 0 fragments.
    Frag a = load_fragA(Xb, m0, 2048, 0, lane);
    Frag b[4];
#pragma unroll
    for (int t = 0; t < 4; ++t) b[t] = load_fragB(Wt, n0 + 16 * t, 2048, 0, lane);

    for (int k0 = 0; k0 < 2048; k0 += 32) {
        // Prefetch next K-step (tail reloads k=0; harmless, keeps EXEC uniform).
        const int kn = (k0 + 32 < 2048) ? (k0 + 32) : 0;
        Frag an = load_fragA(Xb, m0, 2048, kn, lane);
        Frag bn[4];
#pragma unroll
        for (int t = 0; t < 4; ++t) bn[t] = load_fragB(Wt, n0 + 16 * t, 2048, kn, lane);

        // Consume current fragments.
#pragma unroll
        for (int t = 0; t < 4; ++t) acc[t] = wmma_bf16(a, b[t], acc[t]);

        a = an;
#pragma unroll
        for (int t = 0; t < 4; ++t) b[t] = bn[t];
    }

    const int half = (lane >> 4) & 1;
    const int cn   = lane & 15;
#pragma unroll
    for (int r = 0; r < 8; ++r) {
        int m = m0 + r + 8 * half;
#pragma unroll
        for (int t = 0; t < 4; ++t) {
            int n = n0 + 16 * t + cn;
            if (!transposed) {
                C[(size_t)m * 256 + n] = f2bf(acc[t][r]);
            } else {
                int bb = m >> 11;          // batch
                int l  = m & 2047;         // sequence pos
                C[(size_t)bb * 524288 + (size_t)n * 2048 + l] = f2bf(acc[t][r]);
            }
        }
    }
}

// ---------------------------------------------------------------------------
// Kernel 4: causal flash attention, split along keys (flash-decoding style).
// One wave per (batch, 16-query tile, 512-key segment). Q fragments are held
// resident in registers across the whole key loop. Writes unnormalized
// partial O and per-row (m, l) stats for the combine kernel.
// Qb, Kb: [8192 x 256] bf16 row-major. Vt: per batch [256 x 2048] bf16.
// ---------------------------------------------------------------------------
__global__ void flash_partial(const u16* __restrict__ Qb, const u16* __restrict__ Kb,
                              const u16* __restrict__ Vt,
                              float* __restrict__ Opart, float* __restrict__ stats) {
    __shared__ u16 smem[8 * 512];          // 1KB (16x32 bf16) per wave
    const int lane = threadIdx.x & 31;
    const int warp = threadIdx.x >> 5;
    const int widx = blockIdx.x * 8 + warp;   // 0..2047
    const int seg  = widx & 3;                // key segment (512 keys each)
    const int tq   = widx >> 2;               // 0..511: (batch, qtile)
    const int b    = tq >> 7;
    const int q0   = (tq & 127) * 16;
    const int limit = q0 + 16;                // exclusive key bound (causal)
    const int s0key = seg * 512;
    if (s0key >= limit) return;               // wave-uniform early exit

    const int half = (lane >> 4) & 1;
    const int cn   = lane & 15;
    u16* sp = &smem[warp * 512];

    // Q tile (16 x 256) resident in registers: 8 A-fragments.
    Frag qa[8];
#pragma unroll
    for (int kk = 0; kk < 8; ++kk)
        qa[kk] = load_fragA(Qb, b * 2048 + q0, 256, kk * 32, lane);

    f32x8 o[16];
#pragma unroll
    for (int v = 0; v < 16; ++v) o[v] = zero8();
    float m_r[8], l_r[8];
#pragma unroll
    for (int r = 0; r < 8; ++r) { m_r[r] = -3.0e38f; l_r[r] = 0.0f; }

    const float L2E   = 1.4426950408889634f;
    const float scale = 0.0625f;              // 1/sqrt(256)
    const int key_end = (s0key + 512 < limit) ? (s0key + 512) : limit;
    const int nchunks = (key_end - s0key + 31) >> 5;

    for (int c = 0; c < nchunks; ++c) {
        const int j0 = s0key + c * 32;

        // --- S = Q K^T over d_k = 256 ---
        f32x8 s0 = zero8();
        f32x8 s1 = zero8();
#pragma unroll
        for (int kk = 0; kk < 8; ++kk) {
            Frag kb0 = load_fragB(Kb, b * 2048 + j0,      256, kk * 32, lane);
            Frag kb1 = load_fragB(Kb, b * 2048 + j0 + 16, 256, kk * 32, lane);
            s0 = wmma_bf16(qa[kk], kb0, s0);
            s1 = wmma_bf16(qa[kk], kb1, s1);
        }

        const bool needmask = (j0 + 31 > q0);

        // --- scale, mask, online softmax, stage P to LDS ---
#pragma unroll
        for (int r = 0; r < 8; ++r) {
            float a0 = s0[r] * scale;
            float a1 = s1[r] * scale;
            if (needmask) {
                int qrow = q0 + r + 8 * half;
                if (j0 + cn > qrow)      a0 = -3.0e38f;
                if (j0 + 16 + cn > qrow) a1 = -3.0e38f;
            }
            float t = fmaxf(a0, a1);
            t = fmaxf(t, __shfl_xor(t, 1, 32));
            t = fmaxf(t, __shfl_xor(t, 2, 32));
            t = fmaxf(t, __shfl_xor(t, 4, 32));
            t = fmaxf(t, __shfl_xor(t, 8, 32));
            float mnew = fmaxf(m_r[r], t);
            float corr = exp2f((m_r[r] - mnew) * L2E);
            m_r[r] = mnew;
            float p0 = exp2f((a0 - mnew) * L2E);
            float p1 = exp2f((a1 - mnew) * L2E);
            float rs = p0 + p1;
            rs += __shfl_xor(rs, 1, 32);
            rs += __shfl_xor(rs, 2, 32);
            rs += __shfl_xor(rs, 4, 32);
            rs += __shfl_xor(rs, 8, 32);
            l_r[r] = l_r[r] * corr + rs;

            int mrow = r + 8 * half;
            sp[mrow * 32 + cn]      = f2bf(p0);
            sp[mrow * 32 + 16 + cn] = f2bf(p1);

#pragma unroll
            for (int v = 0; v < 16; ++v) o[v][r] *= corr;
        }

        asm volatile("s_wait_dscnt 0" ::: "memory");

        // --- reload P in A-matrix layout from LDS ---
        Frag pa;
        {
            int prow = cn;
            int pk   = half * 8;
#pragma unroll
            for (int i = 0; i < 8; ++i) {
                pa.us[i]     = sp[prow * 32 + pk + i];
                pa.us[8 + i] = sp[prow * 32 + 16 + pk + i];
            }
        }

        // --- O += P V over 32 keys, full d_v = 256 ---
        const u16* vbase = Vt + (size_t)b * 524288;
#pragma unroll
        for (int v = 0; v < 16; ++v) {
            Frag vb = load_fragB(vbase, v * 16, 2048, j0, lane);
            o[v] = wmma_bf16(pa, vb, o[v]);
        }
    }

    // --- store unnormalized partials + stats ---
#pragma unroll
    for (int r = 0; r < 8; ++r) {
        int idx = (tq * 4 + seg) * 16 + r + 8 * half;
        if (cn == 0) {
            stats[idx]         = m_r[r];
            stats[32768 + idx] = l_r[r];
        }
#pragma unroll
        for (int v = 0; v < 16; ++v) {
            Opart[(size_t)idx * 256 + v * 16 + cn] = o[v][r];
        }
    }
}

// ---------------------------------------------------------------------------
// Kernel 5: merge the key segments: O = sum_s w_s * O_s / sum_s w_s * l_s,
// w_s = exp(m_s - max_s m_s). One block per (batch, qtile).
// ---------------------------------------------------------------------------
__global__ void flash_combine(const float* __restrict__ Opart,
                              const float* __restrict__ stats,
                              float* __restrict__ Out) {
    const int tq = blockIdx.x;            // 0..511
    const int b  = tq >> 7;
    const int q0 = (tq & 127) * 16;
    const int nseg = (q0 + 16 + 511) >> 9;
    const float L2E = 1.4426950408889634f;
    const float* mstat = stats;
    const float* lstat = stats + 32768;

    for (int e = threadIdx.x; e < 16 * 256; e += blockDim.x) {
        int row = e >> 8;
        int col = e & 255;
        float M = -3.0e38f;
        for (int s = 0; s < nseg; ++s)
            M = fmaxf(M, mstat[(tq * 4 + s) * 16 + row]);
        float L = 0.0f, acc = 0.0f;
        for (int s = 0; s < nseg; ++s) {
            int idx = (tq * 4 + s) * 16 + row;
            float w = exp2f((mstat[idx] - M) * L2E);
            L   += w * lstat[idx];
            acc += w * Opart[(size_t)idx * 256 + col];
        }
        Out[(size_t)(b * 2048 + q0 + row) * 256 + col] = acc / L;
    }
}

// ---------------------------------------------------------------------------
extern "C" void kernel_launch(void* const* d_in, const int* in_sizes, int n_in,
                              void* d_out, int out_size, void* d_ws, size_t ws_size,
                              hipStream_t stream) {
    (void)in_sizes; (void)n_in; (void)out_size; (void)ws_size;
    const float* X  = (const float*)d_in[0];
    const float* WQ = (const float*)d_in[1];
    const float* WK = (const float*)d_in[2];
    const float* WV = (const float*)d_in[3];
    float* Out = (float*)d_out;

    // Workspace layout (u16 elements for bf16 tensors)
    u16* Xb  = (u16*)d_ws;                           // 8192*2048  (33.55 MB)
    u16* WtQ = Xb  + (size_t)8192 * 2048;            // 256*2048 each (1 MB)
    u16* WtK = WtQ + (size_t)256 * 2048;
    u16* WtV = WtK + (size_t)256 * 2048;
    u16* Qb  = WtV + (size_t)256 * 2048;             // 8192*256 each (4 MB)
    u16* Kb  = Qb  + (size_t)8192 * 256;
    u16* Vt  = Kb  + (size_t)8192 * 256;             // stored [b][256][2048]

    // Flash partials alias the Xb/Wt region, which is dead after the GEMMs
    // (same stream => ordered). Opart: 2048 waves * 16 * 256 fp32 = 33.55 MB;
    // stats: 2 * 32768 fp32 = 256 KB. Total 33.8 MB < 36.7 MB region.
    float* Opart = (float*)d_ws;
    float* stats = (float*)((char*)d_ws + (size_t)33554432);

    cvt_f32_bf16<<<4096, 256, 0, stream>>>(X, Xb, (8192 * 2048) / 4);
    transpose_w<<<2048, 256, 0, stream>>>(WQ, WtQ);
    transpose_w<<<2048, 256, 0, stream>>>(WK, WtK);
    transpose_w<<<2048, 256, 0, stream>>>(WV, WtV);

    gemm_bf16<<<256, 256, 0, stream>>>(Xb, WtQ, Qb, 0);
    gemm_bf16<<<256, 256, 0, stream>>>(Xb, WtK, Kb, 0);
    gemm_bf16<<<256, 256, 0, stream>>>(Xb, WtV, Vt, 1);

    flash_partial<<<256, 256, 0, stream>>>(Qb, Kb, Vt, Opart, stats);
    flash_combine<<<512, 256, 0, stream>>>(Opart, stats, Out);
}
